// NeighborhoodGNN_69200513073727
// MI455X (gfx1250) — compile-verified, matrix operand order checked by reading, not run on hardware
//
#include <hip/hip_runtime.h>
#include <hip/hip_bf16.h>
#include <math.h>

typedef __attribute__((ext_vector_type(16))) _Float16 v16h;
typedef __attribute__((ext_vector_type(8)))  _Float16 v8h;
typedef __attribute__((ext_vector_type(4)))  _Float16 v4h;
typedef __attribute__((ext_vector_type(8)))  float    v8f;

union V16U { v16h v; v8h h[2]; };

// Load one 16x32 f16 WMMA fragment (A or B-transposed) from an LDS row.
// CDNA5 layout: lane L (0-15): K = {k0..k0+7, k0+16..k0+23};
//               lane L+16   : K = {k0+8..k0+15, k0+24..k0+31}.
static __device__ __forceinline__ v16h load_frag(const _Float16* row, int k0, int lane) {
  V16U u;
  const _Float16* p = row + k0 + ((lane & 16) ? 8 : 0);
  u.h[0] = *(const v8h*)p;
  u.h[1] = *(const v8h*)(p + 16);
  return u.v;
}

// ---------------------------------------------------------------- zero
__global__ void zero_kernel(float* __restrict__ p, size_t n) {
  size_t i = (size_t)blockIdx.x * blockDim.x + threadIdx.x;
  size_t stride = (size_t)gridDim.x * blockDim.x;
  for (; i < n; i += stride) p[i] = 0.0f;
}

// ---------------------------------------------------------------- encoders
// h = relu(node_features(V,14) @ W(14,64) + b) ; block = 4 nodes x 64 cols
__global__ void __launch_bounds__(256)
enc_node_kernel(const float* __restrict__ nf, const float* __restrict__ W,
                const float* __restrict__ b, float* __restrict__ h, int V) {
  __shared__ float sW[14 * 64];
  __shared__ float sB[64];
  __shared__ float sF[4][14];
  const int tid = threadIdx.x;
  for (int i = tid; i < 14 * 64; i += 256) sW[i] = W[i];
  if (tid < 64) sB[tid] = b[tid];
  const int v0 = blockIdx.x * 4;
  if (tid < 56) {
    int r = tid / 14, c = tid % 14;
    int v = v0 + r;
    sF[r][c] = (v < V) ? nf[(size_t)v * 14 + c] : 0.0f;
  }
  __syncthreads();
  const int r = tid >> 6, n = tid & 63;
  const int v = v0 + r;
  if (v < V) {
    float s = sB[n];
#pragma unroll
    for (int k = 0; k < 14; ++k) s += sF[r][k] * sW[k * 64 + n];
    h[(size_t)v * 64 + n] = s > 0.0f ? s : 0.0f;
  }
}

// e = relu(edge_attr(E,4) @ W(4,64) + b) ; block = 4 edges x 64 cols
__global__ void __launch_bounds__(256)
enc_edge_kernel(const float* __restrict__ ea, const float* __restrict__ W,
                const float* __restrict__ b, float* __restrict__ e, int E) {
  __shared__ float sW[4 * 64];
  __shared__ float sB[64];
  __shared__ float sF[4][4];
  const int tid = threadIdx.x;
  if (tid < 256) { if (tid < 4 * 64) sW[tid] = W[tid]; }
  if (tid < 64) sB[tid] = b[tid];
  const int e0 = blockIdx.x * 4;
  if (tid < 16) {
    int r = tid >> 2, c = tid & 3;
    int idx = e0 + r;
    sF[r][c] = (idx < E) ? ea[(size_t)idx * 4 + c] : 0.0f;
  }
  __syncthreads();
  const int r = tid >> 6, n = tid & 63;
  const int idx = e0 + r;
  if (idx < E) {
    float s = sB[n];
#pragma unroll
    for (int k = 0; k < 4; ++k) s += sF[r][k] * sW[k * 64 + n];
    e[(size_t)idx * 64 + n] = s > 0.0f ? s : 0.0f;
  }
}

// ---------------------------------------------------------------- edge update (WMMA)
// m = relu(concat(h[src], h[dst], e) @ edge_W(192,64) + b); e := m; agg[dst] += m
__global__ void __launch_bounds__(128)
edge_kernel(const float* __restrict__ hbuf, float* __restrict__ ebuf,
            float* __restrict__ agg, const int* __restrict__ srcIdx,
            const int* __restrict__ dstIdx, const float* __restrict__ eW,
            const float* __restrict__ eB, int E) {
  __shared__ __align__(16) _Float16 sW[64][200];     // Wt[n][k], k=0..191
  __shared__ __align__(16) _Float16 sA[4][16][200];  // per-wave 16 rows x 192
  __shared__ float sBias[64];
  __shared__ int sSrc[4][16];
  __shared__ int sDst[4][16];

  const int tid = threadIdx.x;
  const int lane = tid & 31;
  const int w = tid >> 5;

  for (int i = tid; i < 192 * 64; i += 128) {
    int k = i >> 6, n = i & 63;
    sW[n][k] = (_Float16)eW[i];
  }
  if (tid < 64) sBias[tid] = eB[tid];
  __syncthreads();

  const int tiles = (E + 15) >> 4;
  for (int t0 = blockIdx.x * 4; t0 < tiles; t0 += gridDim.x * 4) {
    int tile = t0 + w;
    const bool active = tile < tiles;
    if (!active) tile = tiles - 1;  // clamp: keep EXEC uniform for WMMA
    const int base = tile << 4;

    if (lane < 16) {
      int eidx = base + lane;
      if (eidx >= E) eidx = E - 1;
      sSrc[w][lane] = srcIdx[eidx];
      sDst[w][lane] = dstIdx[eidx];
    }
    __syncthreads();

    // Stage A: 16 rows x 192 cols = 768 float4 loads across 32 lanes
    for (int it = 0; it < 24; ++it) {
      int idx = lane + 32 * it;
      int r = idx / 48;
      int c = (idx % 48) << 2;
      const float* p;
      if (c < 64) {
        p = &hbuf[(size_t)sSrc[w][r] * 64 + c];
      } else if (c < 128) {
        p = &hbuf[(size_t)sDst[w][r] * 64 + (c - 64)];
      } else {
        int eidx = base + r;
        if (eidx >= E) eidx = E - 1;
        p = &ebuf[(size_t)eidx * 64 + (c - 128)];
      }
      float4 fv = *(const float4*)p;
      v4h q;
      q[0] = (_Float16)fv.x; q[1] = (_Float16)fv.y;
      q[2] = (_Float16)fv.z; q[3] = (_Float16)fv.w;
      *(v4h*)&sA[w][r][c] = q;
    }
    __syncthreads();

    v8f acc[4];
#pragma unroll
    for (int ct = 0; ct < 4; ++ct) {
      float b = sBias[ct * 16 + (lane & 15)];
      v8f t = {b, b, b, b, b, b, b, b};
      acc[ct] = t;
    }
    const _Float16* arow = &sA[w][lane & 15][0];
#pragma unroll
    for (int kt = 0; kt < 6; ++kt) {
      v16h af = load_frag(arow, kt * 32, lane);
#pragma unroll
      for (int ct = 0; ct < 4; ++ct) {
        v16h bf = load_frag(&sW[ct * 16 + (lane & 15)][0], kt * 32, lane);
        acc[ct] = __builtin_amdgcn_wmma_f32_16x16x32_f16(
            false, af, false, bf, (short)0, acc[ct], false, false);
      }
    }

    if (active) {
#pragma unroll
      for (int ct = 0; ct < 4; ++ct) {
        int n = ct * 16 + (lane & 15);
#pragma unroll
        for (int i = 0; i < 8; ++i) {
          int m = i + ((lane & 16) ? 8 : 0);
          int eidx = base + m;
          if (eidx < E) {
            float val = acc[ct][i];
            val = val > 0.0f ? val : 0.0f;
            ebuf[(size_t)eidx * 64 + n] = val;
            atomicAdd(&agg[(size_t)sDst[w][m] * 64 + n], val);
          }
        }
      }
    }
    __syncthreads();
  }
}

// ---------------------------------------------------------------- node update (WMMA)
// h += relu(concat(h, agg) @ node_W(128,64) + b)
__global__ void __launch_bounds__(128)
node_kernel(float* __restrict__ hbuf, const float* __restrict__ agg,
            const float* __restrict__ nW, const float* __restrict__ nB, int V) {
  __shared__ __align__(16) _Float16 sW[64][136];
  __shared__ __align__(16) _Float16 sA[4][16][136];
  __shared__ float sBias[64];

  const int tid = threadIdx.x;
  const int lane = tid & 31;
  const int w = tid >> 5;

  for (int i = tid; i < 128 * 64; i += 128) {
    int k = i >> 6, n = i & 63;
    sW[n][k] = (_Float16)nW[i];
  }
  if (tid < 64) sBias[tid] = nB[tid];
  __syncthreads();

  const int tiles = (V + 15) >> 4;
  for (int t0 = blockIdx.x * 4; t0 < tiles; t0 += gridDim.x * 4) {
    int tile = t0 + w;
    const bool active = tile < tiles;
    if (!active) tile = tiles - 1;
    const int base = tile << 4;

    // Stage: 16 rows x 128 cols = 512 float4
    for (int it = 0; it < 16; ++it) {
      int idx = lane + 32 * it;
      int r = idx >> 5;
      int c = (idx & 31) << 2;
      int v = base + r;
      float4 fv;
      if (v < V) {
        fv = (c < 64) ? *(const float4*)&hbuf[(size_t)v * 64 + c]
                      : *(const float4*)&agg[(size_t)v * 64 + (c - 64)];
      } else {
        fv = make_float4(0.f, 0.f, 0.f, 0.f);
      }
      v4h q;
      q[0] = (_Float16)fv.x; q[1] = (_Float16)fv.y;
      q[2] = (_Float16)fv.z; q[3] = (_Float16)fv.w;
      *(v4h*)&sA[w][r][c] = q;
    }
    __syncthreads();

    v8f acc[4];
#pragma unroll
    for (int ct = 0; ct < 4; ++ct) {
      float b = sBias[ct * 16 + (lane & 15)];
      v8f t = {b, b, b, b, b, b, b, b};
      acc[ct] = t;
    }
    const _Float16* arow = &sA[w][lane & 15][0];
#pragma unroll
    for (int kt = 0; kt < 4; ++kt) {
      v16h af = load_frag(arow, kt * 32, lane);
#pragma unroll
      for (int ct = 0; ct < 4; ++ct) {
        v16h bf = load_frag(&sW[ct * 16 + (lane & 15)][0], kt * 32, lane);
        acc[ct] = __builtin_amdgcn_wmma_f32_16x16x32_f16(
            false, af, false, bf, (short)0, acc[ct], false, false);
      }
    }

    if (active) {
#pragma unroll
      for (int ct = 0; ct < 4; ++ct) {
        int n = ct * 16 + (lane & 15);
#pragma unroll
        for (int i = 0; i < 8; ++i) {
          int m = i + ((lane & 16) ? 8 : 0);
          int v = base + m;
          if (v < V) {
            float val = acc[ct][i];
            val = val > 0.0f ? val : 0.0f;
            hbuf[(size_t)v * 64 + n] += val;
          }
        }
      }
    }
    __syncthreads();
  }
}

// ---------------------------------------------------------------- decode + subset head + pooling (WMMA)
// node_emb = h @ dec_W + b  (-> d_out, + pooled atomics)
// sh = relu(node_emb @ sh1_W + b); logits = sh . sh2_W + sh2_b; scores = sigmoid
__global__ void __launch_bounds__(128)
decode_kernel(const float* __restrict__ hbuf, const float* __restrict__ decW,
              const float* __restrict__ decB, const float* __restrict__ sh1W,
              const float* __restrict__ sh1B, const float* __restrict__ sh2W,
              const float* __restrict__ sh2B, const int* __restrict__ gidx,
              float* __restrict__ out_logits, float* __restrict__ out_scores,
              float* __restrict__ out_emb, float* __restrict__ ge,
              float* __restrict__ cnt, int V) {
  __shared__ __align__(16) _Float16 sWd[64][72];
  __shared__ __align__(16) _Float16 sWs[64][72];
  __shared__ __align__(16) _Float16 sA[4][16][72];
  __shared__ __align__(16) _Float16 sE[4][16][72];
  __shared__ float sBd[64], sBs[64], sS2[64];

  const int tid = threadIdx.x;
  const int lane = tid & 31;
  const int w = tid >> 5;

  for (int i = tid; i < 64 * 64; i += 128) {
    int k = i >> 6, n = i & 63;
    sWd[n][k] = (_Float16)decW[i];
    sWs[n][k] = (_Float16)sh1W[i];
  }
  if (tid < 64) {
    sBd[tid] = decB[tid];
    sBs[tid] = sh1B[tid];
    sS2[tid] = sh2W[tid];
  }
  const float sh2b = sh2B[0];
  __syncthreads();

  const int tiles = (V + 15) >> 4;
  for (int t0 = blockIdx.x * 4; t0 < tiles; t0 += gridDim.x * 4) {
    int tile = t0 + w;
    const bool active = tile < tiles;
    if (!active) tile = tiles - 1;
    const int base = tile << 4;

    // Stage h rows: 16 x 64 = 256 float4
    for (int it = 0; it < 8; ++it) {
      int idx = lane + 32 * it;
      int r = idx >> 4;
      int c = (idx & 15) << 2;
      int v = base + r;
      float4 fv = (v < V) ? *(const float4*)&hbuf[(size_t)v * 64 + c]
                          : make_float4(0.f, 0.f, 0.f, 0.f);
      v4h q;
      q[0] = (_Float16)fv.x; q[1] = (_Float16)fv.y;
      q[2] = (_Float16)fv.z; q[3] = (_Float16)fv.w;
      *(v4h*)&sA[w][r][c] = q;
    }
    __syncthreads();

    // Stage 1: node_emb
    v8f acc[4];
#pragma unroll
    for (int ct = 0; ct < 4; ++ct) {
      float b = sBd[ct * 16 + (lane & 15)];
      v8f t = {b, b, b, b, b, b, b, b};
      acc[ct] = t;
    }
    const _Float16* arow = &sA[w][lane & 15][0];
#pragma unroll
    for (int kt = 0; kt < 2; ++kt) {
      v16h af = load_frag(arow, kt * 32, lane);
#pragma unroll
      for (int ct = 0; ct < 4; ++ct) {
        v16h bf = load_frag(&sWd[ct * 16 + (lane & 15)][0], kt * 32, lane);
        acc[ct] = __builtin_amdgcn_wmma_f32_16x16x32_f16(
            false, af, false, bf, (short)0, acc[ct], false, false);
      }
    }

    // Emit node_emb (no relu), pool atomics, stage f16 copy for stage 2
#pragma unroll
    for (int ct = 0; ct < 4; ++ct) {
      int n = ct * 16 + (lane & 15);
#pragma unroll
      for (int i = 0; i < 8; ++i) {
        int m = i + ((lane & 16) ? 8 : 0);
        float val = acc[ct][i];
        sE[w][m][n] = (_Float16)val;
        int v = base + m;
        if (active && v < V) {
          out_emb[(size_t)v * 64 + n] = val;
          int g = gidx[v];
          atomicAdd(&ge[(size_t)g * 64 + n], val);
          if (ct == 0 && (lane & 15) == 0) atomicAdd(&cnt[g], 1.0f);
        }
      }
    }
    __syncthreads();

    // Stage 2: sh = relu(node_emb @ sh1_W + b)
    v8f acc2[4];
#pragma unroll
    for (int ct = 0; ct < 4; ++ct) {
      float b = sBs[ct * 16 + (lane & 15)];
      v8f t = {b, b, b, b, b, b, b, b};
      acc2[ct] = t;
    }
    const _Float16* erow = &sE[w][lane & 15][0];
#pragma unroll
    for (int kt = 0; kt < 2; ++kt) {
      v16h af = load_frag(erow, kt * 32, lane);
#pragma unroll
      for (int ct = 0; ct < 4; ++ct) {
        v16h bf = load_frag(&sWs[ct * 16 + (lane & 15)][0], kt * 32, lane);
        acc2[ct] = __builtin_amdgcn_wmma_f32_16x16x32_f16(
            false, af, false, bf, (short)0, acc2[ct], false, false);
      }
    }

    // logits = relu(sh) . sh2_W, reduced across the 16 lanes of each half-wave
#pragma unroll
    for (int i = 0; i < 8; ++i) {
      float p = 0.0f;
#pragma unroll
      for (int ct = 0; ct < 4; ++ct) {
        float t = acc2[ct][i];
        t = t > 0.0f ? t : 0.0f;
        p += t * sS2[ct * 16 + (lane & 15)];
      }
      p += __shfl_xor(p, 1);
      p += __shfl_xor(p, 2);
      p += __shfl_xor(p, 4);
      p += __shfl_xor(p, 8);
      int m = i + ((lane & 16) ? 8 : 0);
      int v = base + m;
      if (active && (lane & 15) == 0 && v < V) {
        float lg = p + sh2b;
        out_logits[v] = lg;
        out_scores[v] = 1.0f / (1.0f + __expf(-lg));
      }
    }
    __syncthreads();
  }
}

// ---------------------------------------------------------------- value MLP (tiny, one block)
__global__ void __launch_bounds__(128)
value_kernel(const float* __restrict__ ge, const float* __restrict__ cnt,
             const float* __restrict__ v1W, const float* __restrict__ v1B,
             const float* __restrict__ v2W, const float* __restrict__ v2B,
             const float* __restrict__ v3W, const float* __restrict__ v3B,
             float* __restrict__ out_value) {
  __shared__ float sG[16][64];
  __shared__ float sV1[16][120];
  __shared__ float sV2[16][64];
  const int tid = threadIdx.x;
  for (int i = tid; i < 16 * 64; i += 128) {
    int g = i >> 6, n = i & 63;
    float c = cnt[g];
    c = c < 1.0f ? 1.0f : c;
    sG[g][n] = ge[i] / sqrtf(c);
  }
  __syncthreads();
  for (int i = tid; i < 16 * 120; i += 128) {
    int g = i / 120, n = i % 120;
    float s = v1B[n];
    for (int k = 0; k < 64; ++k) s += sG[g][k] * v1W[k * 120 + n];
    sV1[g][n] = s > 0.0f ? s : 0.0f;
  }
  __syncthreads();
  for (int i = tid; i < 16 * 64; i += 128) {
    int g = i >> 6, n = i & 63;
    float s = v2B[n];
    for (int k = 0; k < 120; ++k) s += sV1[g][k] * v2W[k * 64 + n];
    sV2[g][n] = s > 0.0f ? s : 0.0f;
  }
  __syncthreads();
  if (tid < 16) {
    float s = v3B[0];
    for (int k = 0; k < 64; ++k) s += sV2[tid][k] * v3W[k];
    out_value[tid] = s;
  }
}

// ---------------------------------------------------------------- launch
extern "C" void kernel_launch(void* const* d_in, const int* in_sizes, int n_in,
                              void* d_out, int out_size, void* d_ws, size_t ws_size,
                              hipStream_t stream) {
  (void)n_in; (void)out_size; (void)ws_size;
  const int V = in_sizes[0] / 14;
  const int E = in_sizes[1] / 2;

  const float* node_features  = (const float*)d_in[0];
  const int*   edge_index     = (const int*)d_in[1];
  const float* edge_attr      = (const float*)d_in[2];
  const int*   node_graph_idx = (const int*)d_in[3];
  // d_in[4] = n_graphs (device scalar); reference fixes N_GRAPHS = 16
  const float* enc_n_W = (const float*)d_in[5];
  const float* enc_n_b = (const float*)d_in[6];
  const float* enc_e_W = (const float*)d_in[7];
  const float* enc_e_b = (const float*)d_in[8];
  const float* edge_W  = (const float*)d_in[9];
  const float* edge_b  = (const float*)d_in[10];
  const float* node_W  = (const float*)d_in[11];
  const float* node_b  = (const float*)d_in[12];
  const float* dec_W   = (const float*)d_in[13];
  const float* dec_b   = (const float*)d_in[14];
  const float* sh1_W   = (const float*)d_in[15];
  const float* sh1_b   = (const float*)d_in[16];
  const float* sh2_W   = (const float*)d_in[17];
  const float* sh2_b   = (const float*)d_in[18];
  const float* v1_W    = (const float*)d_in[19];
  const float* v1_b    = (const float*)d_in[20];
  const float* v2_W    = (const float*)d_in[21];
  const float* v2_b    = (const float*)d_in[22];
  const float* v3_W    = (const float*)d_in[23];
  const float* v3_b    = (const float*)d_in[24];

  float* ws    = (float*)d_ws;
  float* h_buf = ws;                                 // V*64
  float* agg   = h_buf + (size_t)V * 64;             // V*64
  float* ge    = agg + (size_t)V * 64;               // 16*64
  float* cnt   = ge + 16 * 64;                       // 16
  float* e_buf = cnt + 16;                           // E*64

  float* out_logits = (float*)d_out;
  float* out_scores = out_logits + V;
  float* out_value  = out_scores + V;
  float* out_emb    = out_value + 16;

  enc_node_kernel<<<(V + 3) / 4, 256, 0, stream>>>(node_features, enc_n_W, enc_n_b, h_buf, V);
  enc_edge_kernel<<<(E + 3) / 4, 256, 0, stream>>>(edge_attr, enc_e_W, enc_e_b, e_buf, E);

  for (int p = 0; p < 5; ++p) {
    zero_kernel<<<2048, 256, 0, stream>>>(agg, (size_t)V * 64);
    edge_kernel<<<2048, 128, 0, stream>>>(h_buf, e_buf, agg, edge_index, edge_index + E,
                                          edge_W, edge_b, E);
    node_kernel<<<1024, 128, 0, stream>>>(h_buf, agg, node_W, node_b, V);
  }

  zero_kernel<<<8, 256, 0, stream>>>(ge, (size_t)(16 * 64 + 16));
  decode_kernel<<<1024, 128, 0, stream>>>(h_buf, dec_W, dec_b, sh1_W, sh1_b, sh2_W, sh2_b,
                                          node_graph_idx, out_logits, out_scores, out_emb,
                                          ge, cnt, V);
  value_kernel<<<1, 128, 0, stream>>>(ge, cnt, v1_W, v1_b, v2_W, v2_b, v3_W, v3_b, out_value);
}
